// TRN_Model_23940147707961
// MI455X (gfx1250) — matrix operand order repaired and checked by reading, not compiled, not run on hardware
//
#include <hip/hip_runtime.h>
#include <stdint.h>

typedef __attribute__((ext_vector_type(16))) _Float16 v16h;
typedef __attribute__((ext_vector_type(8)))  float    v8f;
typedef unsigned int u32x4 __attribute__((ext_vector_type(4)));
typedef int          i32x8 __attribute__((ext_vector_type(8)));
typedef int          i32x4 __attribute__((ext_vector_type(4)));

__device__ __forceinline__ v8f wmma_f16(v16h a, v16h b, v8f c) {
    // D = A(16x32 f16) * B(32x16 f16) + C(16x16 f32)
    return __builtin_amdgcn_wmma_f32_16x16x32_f16(
        false, a, false, b, (short)0, c, false, false);
}

__device__ __forceinline__ float sigmoidf_(float x) {
    return 1.0f / (1.0f + __expf(-x));
}

// ---------------------------------------------------------------------------
// Kernel 0: one-time W_ih f32 -> f16 conversion (1 MB result, L2-resident).
// ---------------------------------------------------------------------------
__global__ void __launch_bounds__(256) cvt_w16(const float* __restrict__ w,
                                               _Float16* __restrict__ w16, int n) {
    int i = blockIdx.x * 256 + threadIdx.x;
    if (i < n) w16[i] = (_Float16)w[i];
}

// ---------------------------------------------------------------------------
// TDM: stream a [256 x 32] f16 tile of W16 (k-chunk kc) into LDS at lds_off.
// D# per CDNA5 ISA 8.3-8.5: group0 = {count=1, lds_addr, global_addr, type=2},
// group1 = {data_size=2B, tensor_dim0=2048, tensor_dim1=256, tile 32x256,
// dim0_stride=2048}; groups 2/3 zero (2D tensor). 6-arg builtin (clang-23).
// ---------------------------------------------------------------------------
__device__ __forceinline__ void tdm_load_wtile(const _Float16* w16, int kc,
                                               uint32_t lds_off) {
    const uint64_t ga = (uint64_t)(uintptr_t)w16 + (uint64_t)kc * 32u * 2u;
    u32x4 g0;
    g0[0] = 1u;                                            // count = 1
    g0[1] = lds_off;                                       // LDS byte address
    g0[2] = (uint32_t)ga;                                  // global_addr[31:0]
    g0[3] = (uint32_t)((ga >> 32) & 0x01FFFFFFu)           // global_addr[56:32]
          | (2u << 30);                                    // type = 2 ("image")
    i32x8 g1;
    g1[0] = (int)(1u << 16);                               // data_size = 1 -> 2 bytes
    g1[1] = (int)(2048u << 16);                            // tensor_dim0[15:0] @63:48
    g1[2] = (int)(256u << 16);                             // dim0[31:16] | dim1[15:0]@95:80
    g1[3] = (int)(32u << 16);                              // dim1[31:16] | tile_dim0@127:112
    g1[4] = 256;                                           // tile_dim1 | tile_dim2<<16
    g1[5] = 2048;                                          // tensor_dim0_stride[31:0]
    g1[6] = 0;                                             // stride0[47:32] | stride1[15:0]
    g1[7] = 0;                                             // stride1[47:16]
    i32x4 g2 = {0, 0, 0, 0};
    i32x4 g3 = {0, 0, 0, 0};
    i32x8 g4 = {0, 0, 0, 0, 0, 0, 0, 0};
    __builtin_amdgcn_tensor_load_to_lds(g0, g1, g2, g3, g4, 0);
}

// ---------------------------------------------------------------------------
// Kernel A: x_proj = x[32768,2048] * W_ih^T + (b_ih + b_hh).
// Wave owns a full 16x256 row strip (16 N-tile accums) -> x read ONCE (HBM
// roofline floor). W16 k-chunks TDM-streamed into TRIPLE-buffered LDS (one
// barrier per k-iter). A operand register-double-buffered (loads for kc+1
// issue under kc's WMMAs). B fragments pipelined TWO tiles deep in two named
// registers so the wait before each WMMA can be a partial dscnt wait.
// ---------------------------------------------------------------------------
__global__ void __launch_bounds__(256) xproj_gemm(
    const float*    __restrict__ x,      // [32768, 2048]
    const _Float16* __restrict__ W16,    // [256, 2048] f16
    const float*    __restrict__ b_ih,   // [256]
    const float*    __restrict__ b_hh,   // [256]
    float*          __restrict__ xp)     // [32768, 256]
{
    __shared__ __attribute__((aligned(32))) _Float16 wtile[3][256 * 32];

    const int lane = threadIdx.x & 31;
    const int wave = threadIdx.x >> 5;
    const int half = lane >> 4;
    const int lr   = lane & 15;
    const int row0 = blockIdx.x * 128 + wave * 16;   // wave's 16 rows
    const int arow = row0 + lr;                      // A row this lane feeds

    v8f acc[16];
    #pragma unroll
    for (int nt = 0; nt < 16; ++nt)
        acc[nt] = (v8f){0.f,0.f,0.f,0.f,0.f,0.f,0.f,0.f};

    // Prologue: A floats for kc=0 (elems 0..15 -> xr[16*half .. 16*half+15])
    const float* xrow = x + (size_t)arow * 2048;
    float4 an[4];
    {
        const float4* p = (const float4*)(xrow + 16 * half);
        an[0] = p[0]; an[1] = p[1]; an[2] = p[2]; an[3] = p[3];
    }
    if (wave == 0)
        tdm_load_wtile(W16, 0, (uint32_t)(uintptr_t)&wtile[0][0]);

    for (int kc = 0; kc < 64; ++kc) {
        if (wave == 0) {
            if (kc + 1 < 64) {
                tdm_load_wtile(W16, kc + 1,
                               (uint32_t)(uintptr_t)&wtile[(kc + 1) % 3][0]);
                __builtin_amdgcn_s_wait_tensorcnt((short)1);  // tile kc done
            } else {
                __builtin_amdgcn_s_wait_tensorcnt((short)0);
            }
        }
        __syncthreads();  // tile kc visible; all waves done with buf (kc-1)%3

        // Convert current A fragment from the prefetched registers.
        v16h a;
        #pragma unroll
        for (int q = 0; q < 4; ++q) {
            a[4 * q + 0] = (_Float16)an[q].x;
            a[4 * q + 1] = (_Float16)an[q].y;
            a[4 * q + 2] = (_Float16)an[q].z;
            a[4 * q + 3] = (_Float16)an[q].w;
        }
        // Issue A loads for kc+1 now; they retire under the 16 WMMAs below.
        if (kc + 1 < 64) {
            const float4* p = (const float4*)(xrow + (kc + 1) * 32 + 16 * half);
            an[0] = p[0]; an[1] = p[1]; an[2] = p[2]; an[3] = p[3];
        }
        if (kc + 2 < 64)
            __builtin_prefetch(xrow + (kc + 2) * 32 + 8 * half, 0, 3);  // WGP scope

        // B fragments pipelined two tiles deep (two live register fragments).
        const _Float16* wt = &wtile[kc % 3][0];
        v16h b0 = *(const v16h*)(wt + (0 * 16 + lr) * 32 + 16 * half);
        v16h b1 = *(const v16h*)(wt + (1 * 16 + lr) * 32 + 16 * half);
        #pragma unroll
        for (int nt = 0; nt < 16; nt += 2) {
            acc[nt] = wmma_f16(a, b0, acc[nt]);
            if (nt + 2 < 16)
                b0 = *(const v16h*)(wt + ((nt + 2) * 16 + lr) * 32 + 16 * half);
            acc[nt + 1] = wmma_f16(a, b1, acc[nt + 1]);
            if (nt + 3 < 16)
                b1 = *(const v16h*)(wt + ((nt + 3) * 16 + lr) * 32 + 16 * half);
        }
    }

    // Bias + store. D layout: VGPR r -> row r+8*half, col = lane&15.
    #pragma unroll
    for (int nt = 0; nt < 16; ++nt) {
        const int col = nt * 16 + lr;
        const float bias = b_ih[col] + b_hh[col];
        #pragma unroll
        for (int r = 0; r < 8; ++r)
            xp[(size_t)(row0 + r + 8 * half) * 256 + col] = acc[nt][r] + bias;
    }
}

// ---------------------------------------------------------------------------
// Kernel B: persistent single-workgroup LSTM scan (T=1024) + final linear.
// Wave w: batch tile m = w&1, h-col tile j = w>>1; gate tiles {j,j+4,j+8,j+12}
// so i/f/g/o land in matching lanes; c-state lives in registers. h double-
// buffered in LDS (f16). x_proj C-init register-double-buffered across steps.
// ---------------------------------------------------------------------------
__global__ void __launch_bounds__(256) lstm_scan(
    const float* __restrict__ xp,     // [1024*32, 256]
    const float* __restrict__ W_hh,   // [256, 64]
    const float* __restrict__ W_lin,  // [101, 64]
    const float* __restrict__ b_lin,  // [101]
    float* __restrict__ out)          // [32, 101]
{
    __shared__ __attribute__((aligned(16))) _Float16 hbuf[2][32 * 64];
    __shared__ float hfin[32 * 64];

    const int lane = threadIdx.x & 31;
    const int wave = threadIdx.x >> 5;
    const int half = lane >> 4;
    const int lr   = lane & 15;
    const int m    = wave & 1;
    const int j    = wave >> 1;
    const int hcol = j * 16 + lr;

    // Preload W_hh^T B-fragments: col g = 64G+16j+lr; elem e -> k = kc*32+16*half+e.
    v16h bf[4][2];
    #pragma unroll
    for (int G = 0; G < 4; ++G)
        #pragma unroll
        for (int kc = 0; kc < 2; ++kc) {
            const float* wr = W_hh + (size_t)(64 * G + 16 * j + lr) * 64
                            + kc * 32 + 16 * half;
            #pragma unroll
            for (int e = 0; e < 16; ++e) bf[G][kc][e] = (_Float16)wr[e];
        }

    float c_st[8];
    #pragma unroll
    for (int r = 0; r < 8; ++r) c_st[r] = 0.f;
    for (int i = threadIdx.x; i < 32 * 64; i += 256) hbuf[0][i] = (_Float16)0.f;

    // Prime the x_proj register buffer with t = 0.
    float nxt[4][8];
    #pragma unroll
    for (int G = 0; G < 4; ++G) {
        const int col = 64 * G + 16 * j + lr;
        #pragma unroll
        for (int r = 0; r < 8; ++r)
            nxt[G][r] = xp[(size_t)(m * 16 + r + 8 * half) * 256 + col];
    }
    __syncthreads();

    for (int t = 0; t < 1024; ++t) {
        v8f acc[4];
        #pragma unroll
        for (int G = 0; G < 4; ++G)
            #pragma unroll
            for (int r = 0; r < 8; ++r) acc[G][r] = nxt[G][r];

        // Issue t+1 loads now; they retire while step t's WMMAs run.
        if (t + 1 < 1024) {
            const size_t b1 = (size_t)(t + 1) * 32 * 256;
            #pragma unroll
            for (int G = 0; G < 4; ++G) {
                const int col = 64 * G + 16 * j + lr;
                #pragma unroll
                for (int r = 0; r < 8; ++r)
                    nxt[G][r] = xp[b1 + (size_t)(m * 16 + r + 8 * half) * 256 + col];
            }
        }
        if (t + 8 < 1024) {
            const size_t pb = (size_t)(t + 8) * 32 * 256
                            + (size_t)(m * 16 + 8 * half) * 256;
            #pragma unroll
            for (int G = 0; G < 4; ++G)
                __builtin_prefetch(&xp[pb + 64 * G + 16 * j + lr], 0, 3);  // WGP scope
        }

        // A fragments from previous h (f16 LDS, double-buffered).
        const _Float16* hb = hbuf[t & 1];
        v16h a[2];
        #pragma unroll
        for (int kc = 0; kc < 2; ++kc) {
            const _Float16* hr = hb + (m * 16 + lr) * 64 + kc * 32;
            #pragma unroll
            for (int e = 0; e < 8; ++e) a[kc][e]     = hr[16 * half + e];
            #pragma unroll
            for (int e = 0; e < 8; ++e) a[kc][8 + e] = hr[16 * half + 8 + e];
        }

        #pragma unroll
        for (int G = 0; G < 4; ++G) {
            acc[G] = wmma_f16(a[0], bf[G][0], acc[G]);
            acc[G] = wmma_f16(a[1], bf[G][1], acc[G]);
        }

        _Float16* hn = hbuf[(t + 1) & 1];
        #pragma unroll
        for (int r = 0; r < 8; ++r) {
            const float iv = sigmoidf_(acc[0][r]);
            const float fv = sigmoidf_(acc[1][r]);
            const float gv = tanhf(acc[2][r]);
            const float ov = sigmoidf_(acc[3][r]);
            const float cn = fv * c_st[r] + iv * gv;
            c_st[r] = cn;
            const float hv = ov * tanhf(cn);
            const int b = m * 16 + r + 8 * half;
            hn[b * 64 + hcol] = (_Float16)hv;
            if (t == 1023) hfin[b * 64 + hcol] = hv;
        }
        __syncthreads();
    }

    for (int idx = threadIdx.x; idx < 32 * 101; idx += 256) {
        const int b  = idx / 101;
        const int cc = idx % 101;
        const float* wr = W_lin + cc * 64;
        const float* hr = hfin + b * 64;
        float s = b_lin[cc];
        #pragma unroll 8
        for (int k = 0; k < 64; ++k) s += hr[k] * wr[k];
        out[idx] = s;
    }
}

extern "C" void kernel_launch(void* const* d_in, const int* in_sizes, int n_in,
                              void* d_out, int out_size, void* d_ws, size_t ws_size,
                              hipStream_t stream) {
    const float* x     = (const float*)d_in[0];  // [1024, 32, 2048]
    const float* W_ih  = (const float*)d_in[1];  // [256, 2048]
    const float* W_hh  = (const float*)d_in[2];  // [256, 64]
    const float* b_ih  = (const float*)d_in[3];  // [256]
    const float* b_hh  = (const float*)d_in[4];  // [256]
    const float* W_lin = (const float*)d_in[5];  // [101, 64]
    const float* b_lin = (const float*)d_in[6];  // [101]
    float* out = (float*)d_out;                  // [32, 101]

    float*    xp  = (float*)d_ws;                            // 32 MiB
    _Float16* w16 = (_Float16*)((char*)d_ws + (size_t)32768 * 256 * 4);  // 1 MiB

    cvt_w16<<<dim3(2048), dim3(256), 0, stream>>>(W_ih, w16, 256 * 2048);
    xproj_gemm<<<dim3(256), dim3(256), 0, stream>>>(x, w16, b_ih, b_hh, xp);
    lstm_scan<<<dim3(1), dim3(256), 0, stream>>>(xp, W_hh, W_lin, b_lin, out);
}
